// Decoder_47210280517794
// MI455X (gfx1250) — matrix operand order repaired and checked by reading, not compiled, not run on hardware
//
#include <hip/hip_runtime.h>
#include <math.h>

#define TPB 256
#define NWG 32

typedef _Float16 half_t;
typedef _Float16 v16h __attribute__((ext_vector_type(16)));
typedef float v8f __attribute__((ext_vector_type(8)));

static constexpr int B_   = 32;
static constexpr int TENC = 256;
static constexpr int MEMD = 640;
static constexpr int HID  = 512;   // ATT_RNN == DEC_RNN
static constexpr int PREN = 256;
static constexpr int ATTD = 128;
static constexpr int NF   = 32;
static constexpr int MEL_ = 80;
static constexpr int SSTEPS = 256;
static constexpr int NGATE = 2048;
static constexpr int KA = 1408;  static constexpr int KBA = KA / 32;  // 44
static constexpr int KD = 1664;  static constexpr int KBD = KD / 32;  // 52
static constexpr int NT = NGATE / 16;                                  // 128
static constexpr float NEGV = -1e20f;

// ---------------- workspace layout (bytes, all 256B aligned) ----------------
static constexpr size_t OFF_CNT  = 0;
static constexpr size_t OFF_AH   = 256;
static constexpr size_t OFF_AC   = OFF_AH   + (size_t)B_*HID*4;
static constexpr size_t OFF_DH   = OFF_AC   + (size_t)B_*HID*4;
static constexpr size_t OFF_DC   = OFF_DH   + (size_t)B_*HID*4;
static constexpr size_t OFF_ACTX = OFF_DC   + (size_t)B_*HID*4;
static constexpr size_t OFF_AW   = OFF_ACTX + (size_t)B_*MEMD*4;
static constexpr size_t OFF_AWC  = OFF_AW   + (size_t)B_*TENC*4;
static constexpr size_t OFF_LA   = OFF_AWC  + (size_t)B_*TENC*4;
static constexpr size_t OFF_GA   = OFF_LA   + (size_t)B_*TENC*4;
static constexpr size_t OFF_GD   = OFF_GA   + (size_t)B_*NGATE*4;
static constexpr size_t OFF_BA   = OFF_GD   + (size_t)B_*NGATE*4;
static constexpr size_t OFF_BD   = OFF_BA   + (size_t)NGATE*4;
static constexpr size_t OFF_XS   = OFF_BD   + (size_t)NGATE*4;
static constexpr size_t OFF_PM   = OFF_XS   + (size_t)SSTEPS*B_*PREN*4;
static constexpr size_t OFF_WA   = OFF_PM   + (size_t)B_*TENC*ATTD*4;
static constexpr size_t OFF_WD   = OFF_WA   + (size_t)NT*KBA*512*2;
static constexpr size_t WS_NEED  = OFF_WD   + (size_t)NT*KBD*512*2;   // ~26.1 MB

__device__ __forceinline__ float sigm(float x){ return 1.f/(1.f+expf(-x)); }

// device-wide barrier: monotonic atomic counter, no reset between phases
__device__ __forceinline__ void gbar(unsigned* cnt, unsigned& tgt){
  __syncthreads();
  tgt += NWG;
  if (threadIdx.x == 0){
    __threadfence();                          // release our stores to L2
    atomicAdd(cnt, 1u);
    while (__hip_atomic_load(cnt, __ATOMIC_ACQUIRE, __HIP_MEMORY_SCOPE_AGENT) < tgt){
      __builtin_amdgcn_s_sleep(1);
    }
  }
  __syncthreads();
  __threadfence();                            // acquire: invalidate L0 before reads
}

// ---------------- one-time setup kernels ----------------
__global__ void k_init(unsigned* cnt, float* AH, float* AC, float* DH, float* DC,
                       float* ACTX, float* AW, float* AWC, float* LA,
                       float* BA, float* BD,
                       const float* abih, const float* abhh,
                       const float* dbih, const float* dbhh){
  int t = blockIdx.x*blockDim.x + threadIdx.x;
  int np = gridDim.x*blockDim.x;
  if (t == 0) *cnt = 0u;
  for (int i=t;i<B_*HID;i+=np){ AH[i]=0.f; AC[i]=0.f; DH[i]=0.f; DC[i]=0.f; }
  for (int i=t;i<B_*MEMD;i+=np) ACTX[i]=0.f;
  for (int i=t;i<B_*TENC;i+=np){ AW[i]=0.f; AWC[i]=0.f; LA[i] = ((i&(TENC-1))==0)?0.f:NEGV; }
  for (int i=t;i<NGATE;i+=np){ BA[i]=abih[i]+abhh[i]; BD[i]=dbih[i]+dbhh[i]; }
}

// prenet for all steps: xs[s,b,:] ; xs[0]=0, xs[s>=1]=prenet(frame 2(s-1)+1)
__global__ void k_prenet(const float* __restrict__ din, const float* __restrict__ w1,
                         const float* __restrict__ w2, float* __restrict__ XS){
  __shared__ float h1[PREN];
  int sb = blockIdx.x;
  int s = sb >> 5, b = sb & 31;
  int tid = threadIdx.x;
  if (s == 0){ XS[(size_t)b*PREN + tid] = 0.f; return; }
  int tcol = 2*(s-1) + 1;
  const float* dcol = din + (size_t)b*MEL_*512 + tcol;   // (B,80,512)
  const float* wr = w1 + (size_t)tid*MEL_;
  float a = 0.f;
  for (int m=0;m<MEL_;++m) a += wr[m]*dcol[(size_t)m*512];
  h1[tid] = fmaxf(a, 0.f);
  __syncthreads();
  const float* w2r = w2 + (size_t)tid*PREN;
  float c = 0.f;
  for (int k=0;k<PREN;++k) c += w2r[k]*h1[k];
  XS[((size_t)s*B_ + b)*PREN + tid] = fmaxf(c, 0.f);
}

// processed_memory[b,t,a] = memory[b,t,:] . memory_w[a,:]
__global__ void k_pm(const float* __restrict__ memory, const float* __restrict__ mw,
                     float* __restrict__ PM){
  int bt = blockIdx.x;
  int a  = threadIdx.x;
  const float* mr = memory + (size_t)bt*MEMD;
  const float* wr = mw + (size_t)a*MEMD;
  float acc = 0.f;
  for (int k=0;k<MEMD;++k) acc += wr[k]*mr[k];
  PM[(size_t)bt*ATTD + a] = acc;
}

// pack combined recurrent weights (wih|whh) into f16 WMMA B-fragments.
// Fragment idx = ((nt*KB + kk)*32 + lane)*16 + i ; lane<16 -> K=kk*32+i, lane>=16 -> K=kk*32+16+i
__global__ void k_packw(const float* __restrict__ awih, const float* __restrict__ awhh,
                        const float* __restrict__ dwih, const float* __restrict__ dwhh,
                        half_t* __restrict__ WA, half_t* __restrict__ WD){
  size_t idx = (size_t)blockIdx.x*blockDim.x + threadIdx.x;
  const size_t totA = (size_t)NT*KBA*512;
  const size_t totD = (size_t)NT*KBD*512;
  if (idx < totA){
    int blk = (int)(idx>>9), sl = (int)(idx&511);
    int nt = blk/KBA, kk = blk%KBA;
    int ln = sl>>4, ii = sl&15;
    int N = nt*16 + (ln&15);
    int K = kk*32 + ((ln&16)?16:0) + ii;
    float v = (K < PREN+MEMD) ? awih[(size_t)N*(PREN+MEMD)+K]
                              : awhh[(size_t)N*HID + (K-(PREN+MEMD))];
    WA[idx] = (half_t)v;
  } else if (idx < totA + totD){
    size_t j = idx - totA;
    int blk = (int)(j>>9), sl = (int)(j&511);
    int nt = blk/KBD, kk = blk%KBD;
    int ln = sl>>4, ii = sl&15;
    int N = nt*16 + (ln&15);
    int K = kk*32 + ((ln&16)?16:0) + ii;
    float v = (K < HID+MEMD) ? dwih[(size_t)N*(HID+MEMD)+K]
                             : dwhh[(size_t)N*HID + (K-(HID+MEMD))];
    WD[j] = (half_t)v;
  }
}

// ---------------- persistent decoder ----------------
struct DecP {
  const float* memory; const int* memlen;
  const float* qw; const float* vw; const float* lconv; const float* llin;
  const float* projw; const float* projb; const float* gatew; const float* gateb;
  unsigned* cnt;
  float *AH,*AC,*DH,*DC,*ACTX,*AW,*AWC,*LA,*GA,*GD,*BA,*BD,*XS,*PM;
  const half_t* WA; const half_t* WD;
  float* out_mel; float* out_gate; float* out_align;
};

// one 16x16 f32 tile per wave; A from LDS, B (weights) from L2-resident pack
__device__ __forceinline__ void gemm_tile(const half_t* Apack, const half_t* W,
                                          const float* bias, float* Gout,
                                          int KB, int wg, int wv, int lane){
  int gw = wg*8 + wv;
  int mt = gw & 1, nt = gw >> 1;
  v8f acc = {0.f,0.f,0.f,0.f,0.f,0.f,0.f,0.f};
  const v16h* Af = (const v16h*)Apack + (size_t)(mt*KB)*32 + lane;
  const v16h* Bf = (const v16h*)W     + (size_t)(nt*KB)*32 + lane;
  for (int kk=0; kk<KB; ++kk){
    v16h av = Af[(size_t)kk*32];
    v16h bv = Bf[(size_t)kk*32];
    acc = __builtin_amdgcn_wmma_f32_16x16x32_f16(false, av, false, bv,
                                                 (short)0, acc, false, false);
  }
  float bb   = bias[nt*16 + (lane&15)];
  int rbase  = mt*16 + ((lane&16)?8:0);
  int col    = nt*16 + (lane&15);
  #pragma unroll
  for (int vi=0; vi<8; ++vi) Gout[(size_t)(rbase+vi)*NGATE + col] = acc[vi] + bb;
}

__global__ __launch_bounds__(TPB) void k_decode(DecP p){
  extern __shared__ char dynsm[];
  half_t* Apack = (half_t*)dynsm;                 // 2*KBD*512 halves = 106496 B max
  __shared__ float s_lconv[NF*62];
  __shared__ float s_llin[ATTD*NF];
  __shared__ float s_vw[ATTD];
  __shared__ float sA[TPB];
  __shared__ float s_aw[TENC], s_awc[TENC], s_la[TENC], s_pq[ATTD];

  const int tid  = threadIdx.x;
  const int wg   = blockIdx.x;
  const int lane = tid & 31;
  const int wv   = tid >> 5;
  unsigned bar = 0;

  for (int i=tid;i<NF*62;i+=TPB)   s_lconv[i] = p.lconv[i];
  for (int i=tid;i<ATTD*NF;i+=TPB) s_llin[i]  = p.llin[i];
  if (tid < ATTD) s_vw[tid] = p.vw[tid];
  __syncthreads();

  for (int s=0; s<SSTEPS; ++s){
    // ---- Phase 1: attention LSTM gates = [x|actx|ah] @ [wih;whh]^T + bias ----
    {
      const int KB = KBA, total = 2*KB*512;
      for (int idx=tid; idx<total; idx+=TPB){
        int mt = idx/(KB*512);
        int r  = idx - mt*(KB*512);
        int kk = r>>9, sl = r&511;
        int ln = sl>>4, ii = sl&15;
        int m  = mt*16 + (ln&15);
        int hi8 = (ln&16)?8:0;
        int k  = kk*32 + 2*hi8 + ii;    // contiguous A-fragment layout (matches B pack)
        float v;
        if (k < PREN)            v = p.XS[((size_t)s*B_+m)*PREN + k];
        else if (k < PREN+MEMD)  v = p.ACTX[m*MEMD + (k-PREN)];
        else                     v = p.AH[m*HID + (k-PREN-MEMD)];
        Apack[idx] = (half_t)v;
      }
      __syncthreads();
      gemm_tile(Apack, p.WA, p.BA, p.GA, KB, wg, wv, lane);
    }
    gbar(p.cnt, bar);

    // ---- Phase 2: per-batch attention (WG b handles batch row b) ----
    {
      const int b = wg;
      for (int n=tid;n<HID;n+=TPB){
        float gi = p.GA[b*NGATE + n];
        float gf = p.GA[b*NGATE + HID + n];
        float gg = p.GA[b*NGATE + 2*HID + n];
        float go = p.GA[b*NGATE + 3*HID + n];
        float c2 = sigm(gf)*p.AC[b*HID+n] + sigm(gi)*tanhf(gg);
        p.AC[b*HID+n] = c2;
        p.AH[b*HID+n] = sigm(go)*tanhf(c2);
      }
      s_aw[tid]  = p.AW[b*TENC+tid];
      s_awc[tid] = p.AWC[b*TENC+tid];
      s_la[tid]  = p.LA[b*TENC+tid];
      __syncthreads();
      if (tid < ATTD){
        const float* q = p.qw + (size_t)tid*HID;
        const float* h = p.AH + (size_t)b*HID;
        float a = 0.f;
        for (int k2=0;k2<HID;++k2) a += q[k2]*h[k2];
        s_pq[tid] = a;
      }
      __syncthreads();
      const int t = tid;
      float lf[NF];
      for (int f=0; f<NF; ++f){
        const float* w0 = s_lconv + f*62;
        float a = 0.f;
        for (int k2=0;k2<31;++k2){
          int tt = t + k2 - 15;
          if (tt>=0 && tt<TENC) a += w0[k2]*s_aw[tt] + w0[31+k2]*s_awc[tt];
        }
        lf[f] = a;
      }
      const float* pmrow = p.PM + ((size_t)b*TENC + t)*ATTD;
      float e = 0.f;
      for (int a2=0;a2<ATTD;++a2){
        const float* ll = s_llin + a2*NF;
        float lo = 0.f;
        #pragma unroll
        for (int f=0;f<NF;++f) lo += ll[f]*lf[f];
        e += s_vw[a2]*tanhf(s_pq[a2] + lo + pmrow[a2]);
      }
      if (t >= p.memlen[b]) e = NEGV;
      float la0 = s_la[t];
      float lsh = (t>0)? s_la[t-1] : NEGV;
      float mx0 = fmaxf(la0,lsh), mn0 = fminf(la0,lsh);
      float lnew = mx0 + log1pf(expf(mn0-mx0)) + e;      // logaddexp + energy
      sA[tid] = lnew; __syncthreads();
      for (int st=TPB/2; st>0; st>>=1){ if(tid<st) sA[tid]=fmaxf(sA[tid],sA[tid+st]); __syncthreads(); }
      float mx = sA[0]; __syncthreads();
      float ex = expf(lnew - mx);
      sA[tid] = ex; __syncthreads();
      for (int st=TPB/2; st>0; st>>=1){ if(tid<st) sA[tid]+=sA[tid+st]; __syncthreads(); }
      float awn = ex / sA[0];
      __syncthreads();
      p.LA[b*TENC+t]  = lnew;
      p.AW[b*TENC+t]  = awn;
      p.AWC[b*TENC+t] = s_awc[t] + awn;
      p.out_align[((size_t)b*SSTEPS + s)*TENC + t] = awn;
      s_aw[t] = awn;
      __syncthreads();
      for (int d=tid; d<MEMD; d+=TPB){
        const float* mcol = p.memory + (size_t)b*TENC*MEMD + d;
        float a = 0.f;
        for (int tt=0; tt<TENC; ++tt) a += s_aw[tt]*mcol[(size_t)tt*MEMD];
        p.ACTX[b*MEMD+d] = a;
      }
    }
    gbar(p.cnt, bar);

    // ---- Phase 3: decoder LSTM gates = [ah|actx|dh] @ [wih;whh]^T + bias ----
    {
      const int KB = KBD, total = 2*KB*512;
      for (int idx=tid; idx<total; idx+=TPB){
        int mt = idx/(KB*512);
        int r  = idx - mt*(KB*512);
        int kk = r>>9, sl = r&511;
        int ln = sl>>4, ii = sl&15;
        int m  = mt*16 + (ln&15);
        int hi8 = (ln&16)?8:0;
        int k  = kk*32 + 2*hi8 + ii;
        float v;
        if (k < HID)            v = p.AH[m*HID + k];
        else if (k < HID+MEMD)  v = p.ACTX[m*MEMD + (k-HID)];
        else                    v = p.DH[m*HID + (k-HID-MEMD)];
        Apack[idx] = (half_t)v;
      }
      __syncthreads();
      gemm_tile(Apack, p.WD, p.BD, p.GD, KB, wg, wv, lane);
    }
    gbar(p.cnt, bar);

    // ---- Phase 4: decoder LSTM elementwise + projection (per-batch WG) ----
    {
      const int b = wg;
      for (int n=tid;n<HID;n+=TPB){
        float gi = p.GD[b*NGATE+n],       gf = p.GD[b*NGATE+HID+n];
        float gg = p.GD[b*NGATE+2*HID+n], go = p.GD[b*NGATE+3*HID+n];
        float c2 = sigm(gf)*p.DC[b*HID+n] + sigm(gi)*tanhf(gg);
        p.DC[b*HID+n] = c2;
        p.DH[b*HID+n] = sigm(go)*tanhf(c2);
      }
      __syncthreads();
      if (tid <= 2*MEL_){          // 0..159 mel rows, 160 = gate
        int j = tid;
        float acc = (j < 2*MEL_) ? p.projb[j] : p.gateb[0];
        const float* wrow = (j < 2*MEL_) ? (p.projw + (size_t)j*(HID+MEMD)) : p.gatew;
        for (int k2=0;k2<HID+MEMD;++k2){
          float h = (k2<HID) ? p.DH[b*HID+k2] : p.ACTX[b*MEMD + (k2-HID)];
          acc += wrow[k2]*h;
        }
        if (j < 2*MEL_)
          p.out_mel[(size_t)b*MEL_*512 + (size_t)(j%MEL_)*512 + (2*s + j/MEL_)] = acc;
        else
          p.out_gate[b*SSTEPS + s] = acc;
      }
    }
    gbar(p.cnt, bar);
  }
}

// ---------------- host ----------------
extern "C" void kernel_launch(void* const* d_in, const int* in_sizes, int n_in,
                              void* d_out, int out_size, void* d_ws, size_t ws_size,
                              hipStream_t stream){
  (void)in_sizes; (void)n_in; (void)out_size;
  if (ws_size < WS_NEED) return;

  const float* memory = (const float*)d_in[0];
  const float* dinp   = (const float*)d_in[1];
  const int*   memlen = (const int*)  d_in[2];
  const float* pw1    = (const float*)d_in[3];
  const float* pw2    = (const float*)d_in[4];
  const float* awih   = (const float*)d_in[5];
  const float* awhh   = (const float*)d_in[6];
  const float* abih   = (const float*)d_in[7];
  const float* abhh   = (const float*)d_in[8];
  const float* qw     = (const float*)d_in[9];
  const float* mw     = (const float*)d_in[10];
  const float* vw     = (const float*)d_in[11];
  const float* lconv  = (const float*)d_in[12];
  const float* llin   = (const float*)d_in[13];
  const float* dwih   = (const float*)d_in[14];
  const float* dwhh   = (const float*)d_in[15];
  const float* dbih   = (const float*)d_in[16];
  const float* dbhh   = (const float*)d_in[17];
  const float* projw  = (const float*)d_in[18];
  const float* projb  = (const float*)d_in[19];
  const float* gatew  = (const float*)d_in[20];
  const float* gateb  = (const float*)d_in[21];

  char* ws = (char*)d_ws;
  unsigned* cnt = (unsigned*)(ws + OFF_CNT);
  float* AH   = (float*)(ws + OFF_AH);
  float* AC   = (float*)(ws + OFF_AC);
  float* DH   = (float*)(ws + OFF_DH);
  float* DC   = (float*)(ws + OFF_DC);
  float* ACTX = (float*)(ws + OFF_ACTX);
  float* AW   = (float*)(ws + OFF_AW);
  float* AWC  = (float*)(ws + OFF_AWC);
  float* LA   = (float*)(ws + OFF_LA);
  float* GA   = (float*)(ws + OFF_GA);
  float* GD   = (float*)(ws + OFF_GD);
  float* BA   = (float*)(ws + OFF_BA);
  float* BD   = (float*)(ws + OFF_BD);
  float* XS   = (float*)(ws + OFF_XS);
  float* PM   = (float*)(ws + OFF_PM);
  half_t* WA  = (half_t*)(ws + OFF_WA);
  half_t* WD  = (half_t*)(ws + OFF_WD);

  float* out_mel   = (float*)d_out;                       // (32,80,512)
  float* out_gate  = out_mel  + (size_t)B_*MEL_*512;      // (32,256)
  float* out_align = out_gate + (size_t)B_*SSTEPS;        // (32,256,256)

  k_init<<<64, 256, 0, stream>>>(cnt, AH,AC,DH,DC,ACTX,AW,AWC,LA,BA,BD,
                                 abih,abhh,dbih,dbhh);
  k_prenet<<<SSTEPS*B_, 256, 0, stream>>>(dinp, pw1, pw2, XS);
  k_pm<<<B_*TENC, 128, 0, stream>>>(memory, mw, PM);
  {
    size_t tot = (size_t)NT*KBA*512 + (size_t)NT*KBD*512;
    k_packw<<<(unsigned)((tot + 255)/256), 256, 0, stream>>>(awih,awhh,dwih,dwhh,WA,WD);
  }

  DecP p;
  p.memory = memory; p.memlen = memlen;
  p.qw = qw; p.vw = vw; p.lconv = lconv; p.llin = llin;
  p.projw = projw; p.projb = projb; p.gatew = gatew; p.gateb = gateb;
  p.cnt = cnt;
  p.AH=AH; p.AC=AC; p.DH=DH; p.DC=DC; p.ACTX=ACTX; p.AW=AW; p.AWC=AWC; p.LA=LA;
  p.GA=GA; p.GD=GD; p.BA=BA; p.BD=BD; p.XS=XS; p.PM=PM;
  p.WA=WA; p.WD=WD;
  p.out_mel=out_mel; p.out_gate=out_gate; p.out_align=out_align;

  const unsigned dynLds = 2u*KBD*512u*sizeof(half_t);     // 106496 B (< 320KB/WGP)
  k_decode<<<NWG, TPB, dynLds, stream>>>(p);
}
